// AttentionLSTMAutoencoder_84327387890297
// MI455X (gfx1250) — matrix operand order repaired and checked by reading, not compiled, not run on hardware
//
#include <hip/hip_runtime.h>
#include <math.h>

// ---------------------------------------------------------------------------
// CDNA5 / gfx1250 WMMA plumbing (wave32, V_WMMA_F32_16X16X32_F16)
// All WMMA operands are kept in "fragment-linear" storage: lane l's 16 halves
// are contiguous at base + l*16, so a fragment load is a single v16h
// (2 x b128) vector load instead of 16 scalar u16 gathers.
// ---------------------------------------------------------------------------
typedef __attribute__((ext_vector_type(16))) _Float16 v16h;
typedef __attribute__((ext_vector_type(8)))  float    v8f;

static_assert(sizeof(_Float16) == 2, "f16 size");

__device__ __forceinline__ v8f wmma_f16(v16h a, v16h b, v8f c) {
  // D = A(16x32) * B(32x16) + C(16x16), f32 accumulate
  return __builtin_amdgcn_wmma_f32_16x16x32_f16(
      /*neg_a=*/false, a, /*neg_b=*/false, b,
      /*c_mod=*/(short)0, c, /*reuse_a=*/false, /*reuse_b=*/false);
}

// --- gfx1250 async global->LDS copy (ASYNCcnt path, CDNA5 ISA) -----------
// VDST = per-lane LDS byte offset; ADDR = 64-bit global address VGPR pair.
__device__ __forceinline__ void async_g2l_b128(unsigned lds_off, const void* gptr) {
  asm volatile("global_load_async_to_lds_b128 %0, %1, off"
               :
               : "v"(lds_off), "v"(gptr)
               : "memory");
}
__device__ __forceinline__ void wait_async0() {
  asm volatile("s_wait_asynccnt 0x0" ::: "memory");
}
// low 32 bits of a generic LDS pointer == LDS byte offset (flat aperture rule)
__device__ __forceinline__ unsigned lds_off_of(const void* p) {
  return (unsigned)(uintptr_t)p;
}

// --- fragment layout swizzles (per 512-half tile) ---
// A tile 16x32 (MxK), contiguous-k order: lane = m + (k&16) ; e = k&15
__device__ __forceinline__ int aoff(int m, int kl) {
  return ((m + (kl & 16)) << 4) + (kl & 15);
}
// B tile 32x16 (KxN), contiguous-k order: lane = n + (k&16) ; e = k&15
__device__ __forceinline__ int boff(int kl, int nl) {
  return ((nl + (kl & 16)) << 4) + (kl & 15);
}
// C tile 16x16 f32: lane = n + 16*(m>=8) ; reg = m&7 ; stored v8f per lane
__device__ __forceinline__ int coff(int m, int nl) {
  return (((nl + ((m & 8) << 1)) << 3) + (m & 7));
}

// fragment-linear v16h load (LDS or global)
__device__ __forceinline__ v16h load_frag(const _Float16* p) {
  return *(const v16h*)(p + ((threadIdx.x & 31) << 4));
}

// Store C (16x16 f32) to row-major p[m*ld + n].
__device__ __forceinline__ void store_c(float* p, int ld, v8f c) {
  int l = threadIdx.x & 31;
  int n = l & 15;
  int mb = (l >> 4) << 3;
#pragma unroll
  for (int r = 0; r < 8; ++r) p[(size_t)(mb + r) * ld + n] = c[r];
}

__device__ __forceinline__ float sigmoidf_(float z) { return 1.f / (1.f + expf(-z)); }

// Problem constants
#define BATCH 1024
#define SEQ   128
#define FEAT  64
#define HID   100
#define NG    400      // 4*HID gates
#define TOK   (BATCH * SEQ)

// ---------------------------------------------------------------------------
// Kernel 0: weight prep — emit f16 B-operand buffers in fragment-tile order
// (chunk-major: tile index = kc*(N/16)+tj, 512 halves per tile), plus biases.
// ---------------------------------------------------------------------------
__device__ void emit_bfrag(const float* __restrict__ W, int N, int K, int Kpad,
                           _Float16* __restrict__ dst, int tid, int nt) {
  int TN = N >> 4;
  int total = Kpad * N;
  for (int i = tid; i < total; i += nt) {
    int kc = i / (TN * 512);
    int rem = i - kc * (TN * 512);
    int tj = rem >> 9;
    int q = rem & 511;
    int lane = q >> 4, e = q & 15;
    int k = kc * 32 + (lane & 16) + e;
    int n = tj * 16 + (lane & 15);
    dst[i] = (_Float16)((k < K) ? W[(size_t)n * K + k] : 0.f);
  }
}

__global__ void prep_kernel(const float* __restrict__ eWih, const float* __restrict__ eWhh,
                            const float* __restrict__ dWhh, const float* __restrict__ oW,
                            const float* __restrict__ fW1, const float* __restrict__ fW2,
                            const float* __restrict__ cW1, const float* __restrict__ cW2,
                            const float* __restrict__ ebih, const float* __restrict__ ebhh,
                            const float* __restrict__ dbih, const float* __restrict__ dbhh,
                            _Float16* eWihT, _Float16* eWhhT, _Float16* dWhhT, _Float16* oWT,
                            _Float16* fW1T, _Float16* fW2T, _Float16* cW1T, _Float16* cW2T,
                            float* ebias, float* dbias) {
  int tid = blockIdx.x * blockDim.x + threadIdx.x;
  int nt = gridDim.x * blockDim.x;
  emit_bfrag(eWih, NG, 64, 64, eWihT, tid, nt);     // [64][400]
  emit_bfrag(eWhh, NG, HID, 128, eWhhT, tid, nt);   // [128][400] padded
  emit_bfrag(dWhh, NG, HID, 128, dWhhT, tid, nt);   // [128][400] padded
  emit_bfrag(oW, 64, HID, 128, oWT, tid, nt);       // [128][64]  padded
  emit_bfrag(fW1, 32, 64, 64, fW1T, tid, nt);       // [64][32]
  emit_bfrag(fW2, 64, 32, 32, fW2T, tid, nt);       // [32][64]
  emit_bfrag(cW1, 64, 96, 96, cW1T, tid, nt);       // [96][64]
  emit_bfrag(cW2, 64, 64, 64, cW2T, tid, nt);       // [64][64]
  for (int i = tid; i < NG; i += nt) { ebias[i] = ebih[i] + ebhh[i]; dbias[i] = dbih[i] + dbhh[i]; }
}

// ---------------------------------------------------------------------------
// Kernel 1: Gram matrix  gram[b] = x_b^T (64x128) @ x_b (128x64)   (WMMA)
// x staged once into LDS in both A-fragment (x^T) and B-fragment (x) order.
// block = 128 threads (4 waves), wave w owns row-tile w; grid = BATCH
// ---------------------------------------------------------------------------
__global__ void gram_kernel(const float* __restrict__ x, float* __restrict__ gram) {
  __shared__ __align__(32) _Float16 xa[16 * 512];   // A tiles (ti,kc): idx=(ti*4+kc)*512
  __shared__ __align__(32) _Float16 xbv[16 * 512];  // B tiles (kc,tj): idx=(kc*4+tj)*512
  int b = blockIdx.x;
  const float* xb = x + (size_t)b * SEQ * FEAT;
  for (int i = threadIdx.x; i < SEQ * FEAT; i += blockDim.x) {
    int s = i >> 6, f = i & 63;
    _Float16 v = (_Float16)xb[i];
    xa[((f >> 4) * 4 + (s >> 5)) * 512 + aoff(f & 15, s & 31)] = v;   // A[m=f,k=s]
    xbv[((s >> 5) * 4 + (f >> 4)) * 512 + boff(s & 31, f & 15)] = v;  // B[k=s,n=f]
  }
  __syncthreads();
  int ti = threadIdx.x >> 5;  // 0..3
  float* g = gram + (size_t)b * FEAT * FEAT;
  for (int tj = 0; tj < 4; ++tj) {
    v8f acc = {0.f, 0.f, 0.f, 0.f, 0.f, 0.f, 0.f, 0.f};
    for (int kc = 0; kc < 4; ++kc) {
      acc = wmma_f16(load_frag(xa + (ti * 4 + kc) * 512),
                     load_frag(xbv + (kc * 4 + tj) * 512), acc);
    }
    store_c(g + ti * 16 * FEAT + tj * 16, FEAT, acc);
  }
}

// ---------------------------------------------------------------------------
// Kernel 2: conv1 (1->16ch, 3x3 SAME) + ReLU + maxpool2 -> pool1[b][16][32][32]
// ---------------------------------------------------------------------------
__global__ void conv1_kernel(const float* __restrict__ gram, const float* __restrict__ w,
                             const float* __restrict__ bias, float* __restrict__ pool1) {
  __shared__ float simg[64 * 64];
  __shared__ float sw[16 * 9];
  __shared__ float sb[16];
  int b = blockIdx.x;
  const float* img = gram + (size_t)b * 4096;
  for (int i = threadIdx.x; i < 4096; i += blockDim.x) simg[i] = img[i];
  for (int i = threadIdx.x; i < 144; i += blockDim.x) sw[i] = w[i];
  if (threadIdx.x < 16) sb[threadIdx.x] = bias[threadIdx.x];
  __syncthreads();
  for (int idx = threadIdx.x; idx < 16 * 32 * 32; idx += blockDim.x) {
    int o = idx >> 10, py = (idx >> 5) & 31, px = idx & 31;
    float mx = -3.4e38f;
    for (int dy = 0; dy < 2; ++dy)
      for (int dx = 0; dx < 2; ++dx) {
        int y = py * 2 + dy, xx = px * 2 + dx;
        float s = sb[o];
        for (int ky = -1; ky <= 1; ++ky)
          for (int kx = -1; kx <= 1; ++kx) {
            int yy = y + ky, xc = xx + kx;
            if (yy >= 0 && yy < 64 && xc >= 0 && xc < 64)
              s += sw[o * 9 + (ky + 1) * 3 + (kx + 1)] * simg[yy * 64 + xc];
          }
        s = s > 0.f ? s : 0.f;
        mx = s > mx ? s : mx;
      }
    pool1[(size_t)b * 16384 + idx] = mx;
  }
}

// ---------------------------------------------------------------------------
// Kernel 3: conv2 (16->32ch) + ReLU + global avg pool -> proc[b][32]
// Deterministic: 8 threads per out-channel, fixed-order LDS tree reduce.
// ---------------------------------------------------------------------------
__global__ void conv2_kernel(const float* __restrict__ pool1, const float* __restrict__ w,
                             const float* __restrict__ bias, float* __restrict__ proc) {
  __shared__ float sw[32 * 16 * 9];
  __shared__ float red[256];
  int b = blockIdx.x;
  const float* in = pool1 + (size_t)b * 16384;
  for (int i = threadIdx.x; i < 4608; i += blockDim.x) sw[i] = w[i];
  __syncthreads();
  int o = threadIdx.x >> 3;   // 0..31
  int sub = threadIdx.x & 7;  // 0..7
  float acc = 0.f;
  for (int p = sub; p < 1024; p += 8) {
    int y = p >> 5, xx = p & 31;
    float s = bias[o];
    for (int ci = 0; ci < 16; ++ci) {
      const float* ip = in + ci * 1024;
      const float* wp = sw + (o * 16 + ci) * 9;
      for (int ky = -1; ky <= 1; ++ky) {
        int yy = y + ky;
        if (yy < 0 || yy >= 32) continue;
        for (int kx = -1; kx <= 1; ++kx) {
          int xc = xx + kx;
          if (xc < 0 || xc >= 32) continue;
          s += wp[(ky + 1) * 3 + (kx + 1)] * ip[yy * 32 + xc];
        }
      }
    }
    acc += (s > 0.f ? s : 0.f);
  }
  red[threadIdx.x] = acc;
  __syncthreads();
  for (int off = 4; off > 0; off >>= 1) {
    if (sub < off) red[threadIdx.x] += red[threadIdx.x + off];
    __syncthreads();
  }
  if (sub == 0) proc[(size_t)b * 32 + o] = red[threadIdx.x] * (1.f / 1024.f);
}

// ---------------------------------------------------------------------------
// Kernel 4: fused token pipeline (WMMA), all operands fragment-linear:
//   fs = tanh(x@fW1^T+fb1)@fW2^T+fb2 ; comb=[fs,proc] ;
//   aw = tanh(comb@cW1^T+cb1)@cW2^T+cb2 ; feature_w = softmax(aw, feat) ;
//   wx = x * feature_w.   block = 128 thr (4 waves x 16-token tiles)
// ---------------------------------------------------------------------------
__global__ void feature_kernel(const float* __restrict__ x, const float* __restrict__ proc,
                               const _Float16* __restrict__ fW1T, const float* __restrict__ fb1,
                               const _Float16* __restrict__ fW2T, const float* __restrict__ fb2,
                               const _Float16* __restrict__ cW1T, const float* __restrict__ cb1,
                               const _Float16* __restrict__ cW2T, const float* __restrict__ cb2,
                               float* __restrict__ fw_out, float* __restrict__ wx) {
  __shared__ __align__(32) _Float16 xwbuf[4 * 2 * 512];   // per-wave x tile, 2 k-chunks
  __shared__ __align__(32) _Float16 t1buf[4 * 1 * 512];   // per-wave, K=32
  __shared__ __align__(32) _Float16 combbuf[4 * 3 * 512]; // per-wave, K=96
  __shared__ __align__(32) _Float16 ubuf[4 * 2 * 512];    // per-wave, K=64
  __shared__ float awbuf[4 * 16 * 64];

  int wv = threadIdx.x >> 5;
  int l  = threadIdx.x & 31;
  size_t tok0 = (size_t)blockIdx.x * 64 + (size_t)wv * 16;
  const float* xt = x + tok0 * FEAT;
  _Float16* xw  = xwbuf + wv * 2 * 512;
  _Float16* t1  = t1buf + wv * 512;
  _Float16* cmb = combbuf + wv * 3 * 512;
  _Float16* u   = ubuf + wv * 2 * 512;
  float*    awb = awbuf + wv * 16 * 64;

  // stage x tile (A-fragment order)
  for (int i = l; i < 16 * 64; i += 32) {
    int r = i >> 6, j = i & 63;
    xw[(j >> 5) * 512 + aoff(r, j & 31)] = (_Float16)xt[i];
  }
  int nl = l & 15, mb = (l >> 4) << 3;
  // t1 = tanh(x @ fW1^T + fb1)  [16x32]
  for (int tj = 0; tj < 2; ++tj) {
    v8f acc = {0.f, 0.f, 0.f, 0.f, 0.f, 0.f, 0.f, 0.f};
    for (int kc = 0; kc < 2; ++kc)
      acc = wmma_f16(load_frag(xw + kc * 512), load_frag(fW1T + (kc * 2 + tj) * 512), acc);
    int n = tj * 16 + nl;
    float bb = fb1[n];
#pragma unroll
    for (int r = 0; r < 8; ++r) t1[aoff(mb + r, n)] = (_Float16)tanhf(acc[r] + bb);
  }
  // fs = t1 @ fW2^T + fb2 -> comb chunks 0..1
  for (int tj = 0; tj < 4; ++tj) {
    v8f acc = {0.f, 0.f, 0.f, 0.f, 0.f, 0.f, 0.f, 0.f};
    acc = wmma_f16(load_frag(t1), load_frag(fW2T + tj * 512), acc);
    int n = tj * 16 + nl;
    float bb = fb2[n];
#pragma unroll
    for (int r = 0; r < 8; ++r) cmb[(n >> 5) * 512 + aoff(mb + r, n & 31)] = (_Float16)(acc[r] + bb);
  }
  // comb chunk 2 = proc[batch(token)]  (k = 64..95)
  for (int i = l; i < 16 * 32; i += 32) {
    int r = i >> 5, j = i & 31;
    size_t tk = tok0 + r;
    int bb = (int)(tk >> 7);
    cmb[2 * 512 + aoff(r, j)] = (_Float16)proc[(size_t)bb * 32 + j];
  }
  // u = tanh(comb @ cW1^T + cb1)  [16x64], K = 96
  for (int tj = 0; tj < 4; ++tj) {
    v8f acc = {0.f, 0.f, 0.f, 0.f, 0.f, 0.f, 0.f, 0.f};
    for (int kc = 0; kc < 3; ++kc)
      acc = wmma_f16(load_frag(cmb + kc * 512), load_frag(cW1T + (kc * 4 + tj) * 512), acc);
    int n = tj * 16 + nl;
    float bb = cb1[n];
#pragma unroll
    for (int r = 0; r < 8; ++r) u[(n >> 5) * 512 + aoff(mb + r, n & 31)] = (_Float16)tanhf(acc[r] + bb);
  }
  // aw = u @ cW2^T + cb2  [16x64]
  for (int tj = 0; tj < 4; ++tj) {
    v8f acc = {0.f, 0.f, 0.f, 0.f, 0.f, 0.f, 0.f, 0.f};
    for (int kc = 0; kc < 2; ++kc)
      acc = wmma_f16(load_frag(u + kc * 512), load_frag(cW2T + (kc * 4 + tj) * 512), acc);
    int n = tj * 16 + nl;
    float bb = cb2[n];
#pragma unroll
    for (int r = 0; r < 8; ++r) awb[(mb + r) * 64 + n] = acc[r] + bb;
  }
  // softmax over features per row; emit feature_w and wx
  if (l < 16) {
    float* row = awb + l * 64;
    float mx = row[0];
    for (int j = 1; j < 64; ++j) mx = fmaxf(mx, row[j]);
    float sm = 0.f;
    for (int j = 0; j < 64; ++j) { float e = expf(row[j] - mx); row[j] = e; sm += e; }
    float inv = 1.f / sm;
    size_t tk = tok0 + l;
    const float* xr = x + tk * FEAT;
    float* fo = fw_out + tk * FEAT;
    float* wr = wx + tk * FEAT;
    for (int j = 0; j < 64; ++j) { float fw = row[j] * inv; fo[j] = fw; wr[j] = xr[j] * fw; }
  }
}

// ---------------------------------------------------------------------------
// Kernel 5: persistent LSTM recurrence (WMMA). One block = 16 batch rows,
// all 128 steps. Whh^T (f16 fragment order, K padded 100->128) brought into
// LDS with gfx1250 async global->LDS B128 copies (102 KB; CDNA5 320KB/WGP).
// Gate matrix kept in C-fragment order so each wave writes its accumulator
// as one contiguous v8f.
// MODE 0: encoder (x-seq + Wih). MODE 1: decoder (constant gate base).
// ---------------------------------------------------------------------------
template <int MODE>
__global__ void lstm_kernel(const float* __restrict__ wxin,       // [TOK][64]    (MODE 0)
                            const _Float16* __restrict__ WihT,    // frag [64][400]  (MODE 0)
                            const _Float16* __restrict__ WhhT,    // frag [128][400]
                            const float* __restrict__ bias,       // [400]        (MODE 0)
                            const float* __restrict__ zbase,      // [B][400]     (MODE 1)
                            const float* __restrict__ h0,         // [B][100]     (MODE 1)
                            const float* __restrict__ c0,         // [B][100]     (MODE 1)
                            float* __restrict__ outH) {           // [TOK][100]
  __shared__ __align__(32) _Float16 sWhh[128 * NG];   // 102400 B, frag tiles (kc*25+t)
  __shared__ __align__(32) _Float16 xb16[2 * 512];    //   2048 B
  __shared__ __align__(32) _Float16 hb[4 * 512];      //   4096 B (k=100..127 stays 0)
  __shared__ __align__(32) float    zb[25 * 256];     //  25600 B, C-fragment order
  __shared__ float cbuf[16 * HID];                    //   6400 B

  int b0 = blockIdx.x * 16;
  // async DMA of Whh^T image into LDS (6400 x 16B chunks, ASYNCcnt-tracked)
  {
    unsigned lbase = lds_off_of(&sWhh[0]);
    const char* gbase = (const char*)WhhT;
    for (int c = threadIdx.x; c < (128 * NG) / 8; c += blockDim.x)
      async_g2l_b128(lbase + (unsigned)c * 16u, gbase + (size_t)c * 16u);
  }
  for (int i = threadIdx.x; i < 4 * 512; i += blockDim.x) hb[i] = (_Float16)0.f;
  for (int i = threadIdx.x; i < 16 * HID; i += blockDim.x) cbuf[i] = 0.f;
  if (MODE == 1) {
    __syncthreads();
    for (int i = threadIdx.x; i < 16 * HID; i += blockDim.x) {
      int r = i / HID, j = i % HID;
      hb[(j >> 5) * 512 + aoff(r, j & 31)] = (_Float16)h0[(size_t)(b0 + r) * HID + j];
      cbuf[i] = c0[(size_t)(b0 + r) * HID + j];
    }
  }
  wait_async0();
  __syncthreads();

  int l = threadIdx.x & 31;
  for (int s = 0; s < SEQ; ++s) {
    if (MODE == 0) {
      for (int i = threadIdx.x; i < 16 * 64; i += blockDim.x) {
        int r = i >> 6, j = i & 63;
        xb16[(j >> 5) * 512 + aoff(r, j & 31)] =
            (_Float16)wxin[((size_t)(b0 + r) * SEQ + s) * FEAT + j];
      }
      // warm caches for the next step's x slice (global_prefetch_b8)
      if (threadIdx.x < 16 && s + 1 < SEQ)
        __builtin_prefetch(&wxin[((size_t)(b0 + threadIdx.x) * SEQ + (s + 1)) * FEAT], 0, 3);
    }
    __syncthreads();
    // z tiles: 25 n-tiles of 16 across 8 waves
    for (int t = (threadIdx.x >> 5); t < 25; t += 8) {
      v8f acc = {0.f, 0.f, 0.f, 0.f, 0.f, 0.f, 0.f, 0.f};
      if (MODE == 0) {
        for (int kc = 0; kc < 2; ++kc)   // x @ Wih^T, K = 64
          acc = wmma_f16(load_frag(xb16 + kc * 512),
                         load_frag(WihT + (kc * 25 + t) * 512), acc);
      }
      for (int kc = 0; kc < 4; ++kc)     // h @ Whh^T, K = 128 (padded)
        acc = wmma_f16(load_frag(hb + kc * 512),
                       load_frag(sWhh + (kc * 25 + t) * 512), acc);
      *(v8f*)(zb + t * 256 + l * 8) = acc;   // contiguous C-fragment store
    }
    __syncthreads();
    // gate nonlinearity + state update (torch gate order i,f,g,o)
    for (int i = threadIdx.x; i < 16 * HID; i += blockDim.x) {
      int r = i / HID, j = i % HID;
      size_t brow = (size_t)(b0 + r);
      int n0 = j, n1 = HID + j, n2 = 2 * HID + j, n3 = 3 * HID + j;
      float zi = zb[(n0 >> 4) * 256 + coff(r, n0 & 15)];
      float zf = zb[(n1 >> 4) * 256 + coff(r, n1 & 15)];
      float zg = zb[(n2 >> 4) * 256 + coff(r, n2 & 15)];
      float zo = zb[(n3 >> 4) * 256 + coff(r, n3 & 15)];
      if (MODE == 0) {
        zi += bias[n0]; zf += bias[n1]; zg += bias[n2]; zo += bias[n3];
      } else {
        const float* zr = zbase + brow * NG;
        zi += zr[n0]; zf += zr[n1]; zg += zr[n2]; zo += zr[n3];
      }
      float ig = sigmoidf_(zi), fg = sigmoidf_(zf), gg = tanhf(zg), og = sigmoidf_(zo);
      float c = fg * cbuf[i] + ig * gg;
      float h = og * tanhf(c);
      cbuf[i] = c;
      hb[(j >> 5) * 512 + aoff(r, j & 31)] = (_Float16)h;
      outH[(brow * SEQ + s) * HID + j] = h;
    }
    __syncthreads();
  }
}

// ---------------------------------------------------------------------------
// Kernel 6: temporal score per token: ts = tanh(enc@aW1^T+ab1)@aW2^T+ab2
// ---------------------------------------------------------------------------
__global__ void tscore_kernel(const float* __restrict__ enc, const float* __restrict__ aW1,
                              const float* __restrict__ ab1, const float* __restrict__ aW2,
                              const float* __restrict__ ab2, float* __restrict__ ts) {
  size_t t = (size_t)blockIdx.x * blockDim.x + threadIdx.x;
  if (t >= TOK) return;
  const float* e = enc + t * HID;
  float acc = ab2[0];
  for (int j = 0; j < 64; ++j) {
    float d = ab1[j];
    const float* wr = aW1 + j * HID;
    for (int k = 0; k < HID; ++k) d += wr[k] * e[k];
    acc += aW2[j] * tanhf(d);
  }
  ts[t] = acc;
}

// ---------------------------------------------------------------------------
// Kernel 7: temporal softmax (over S) + context  ctx[b] = sum_s tw * enc
// block = 128 threads = one batch row
// ---------------------------------------------------------------------------
__global__ void tctx_kernel(const float* __restrict__ ts, const float* __restrict__ enc,
                            float* __restrict__ time_w, float* __restrict__ ctx) {
  __shared__ float sv[SEQ];
  __shared__ float red[SEQ];
  int b = blockIdx.x;
  int s = threadIdx.x;
  float v = ts[(size_t)b * SEQ + s];
  red[s] = v;
  __syncthreads();
  for (int o = 64; o > 0; o >>= 1) { if (s < o) red[s] = fmaxf(red[s], red[s + o]); __syncthreads(); }
  float mx = red[0];
  __syncthreads();
  float e = expf(v - mx);
  red[s] = e;
  __syncthreads();
  for (int o = 64; o > 0; o >>= 1) { if (s < o) red[s] += red[s + o]; __syncthreads(); }
  float w = e / red[0];
  sv[s] = w;
  time_w[(size_t)b * SEQ + s] = w;
  __syncthreads();
  for (int h = threadIdx.x; h < HID; h += blockDim.x) {
    float acc = 0.f;
    for (int s2 = 0; s2 < SEQ; ++s2) acc += sv[s2] * enc[((size_t)b * SEQ + s2) * HID + h];
    ctx[(size_t)b * HID + h] = acc;
  }
}

// ---------------------------------------------------------------------------
// Kernel 8: bottleneck + decoder initial states
// ---------------------------------------------------------------------------
__global__ void bott_kernel(const float* __restrict__ ctx, const float* __restrict__ bW,
                            const float* __restrict__ bb, const float* __restrict__ h0W,
                            const float* __restrict__ h0b, const float* __restrict__ c0W,
                            const float* __restrict__ c0b, float* __restrict__ bott,
                            float* __restrict__ dh0, float* __restrict__ dc0) {
  __shared__ float sb[32];
  int b = blockIdx.x;
  int t = threadIdx.x;
  if (t < 32) {
    float a = bb[t];
    const float* c = ctx + (size_t)b * HID;
    for (int k = 0; k < HID; ++k) a += bW[t * HID + k] * c[k];
    sb[t] = a;
    bott[(size_t)b * 32 + t] = a;
  }
  __syncthreads();
  if (t < HID) {
    float a = h0b[t], a2 = c0b[t];
    for (int k = 0; k < 32; ++k) { float v = sb[k]; a += h0W[t * 32 + k] * v; a2 += c0W[t * 32 + k] * v; }
    dh0[(size_t)b * HID + t] = a;
    dc0[(size_t)b * HID + t] = a2;
  }
}

// ---------------------------------------------------------------------------
// Kernel 9: decoder constant gate base  zbase[b] = bott@dWih^T + dbih + dbhh
// ---------------------------------------------------------------------------
__global__ void dxp_kernel(const float* __restrict__ bott, const float* __restrict__ dWih,
                           const float* __restrict__ dbias, float* __restrict__ zbase) {
  int i = blockIdx.x * blockDim.x + threadIdx.x;
  if (i >= BATCH * NG) return;
  int b = i / NG, n = i % NG;
  float a = dbias[n];
  const float* bt = bott + (size_t)b * 32;
  const float* wr = dWih + (size_t)n * 32;
  for (int k = 0; k < 32; ++k) a += wr[k] * bt[k];
  zbase[i] = a;
}

// ---------------------------------------------------------------------------
// Kernel 10: output projection  out = dec @ oW^T + ob   (WMMA, K=100 pad 128)
// B image async-DMA'd into LDS; per-wave A tile staged in fragment order.
// block = 256 (8 waves x 16-token tiles)
// ---------------------------------------------------------------------------
__global__ void outproj_kernel(const float* __restrict__ dec, const _Float16* __restrict__ oWT,
                               const float* __restrict__ ob, float* __restrict__ out) {
  __shared__ __align__(32) _Float16 sB[16 * 512];       // 16 KB: tiles (kc*4+tj)
  __shared__ __align__(32) _Float16 abuf[8 * 4 * 512];  // 32 KB: per-wave, 4 k-chunks
  {
    unsigned lbase = lds_off_of(&sB[0]);
    const char* gbase = (const char*)oWT;
    for (int c = threadIdx.x; c < (16 * 512) / 8; c += blockDim.x)
      async_g2l_b128(lbase + (unsigned)c * 16u, gbase + (size_t)c * 16u);
  }
  int wv = threadIdx.x >> 5;
  int l = threadIdx.x & 31;
  _Float16* aw = abuf + wv * 4 * 512;
  size_t tok0 = ((size_t)blockIdx.x * 8 + wv) * 16;
  const float* ap = dec + tok0 * HID;
  // zero-pad then stage A tile (k = 100..127 zero)
  for (int i = l; i < 4 * 512; i += 32) aw[i] = (_Float16)0.f;
  for (int i = l; i < 16 * HID; i += 32) {
    int r = i / HID, j = i % HID;
    aw[(j >> 5) * 512 + aoff(r, j & 31)] = (_Float16)ap[i];
  }
  wait_async0();
  __syncthreads();
  int nl = l & 15, mb = (l >> 4) << 3;
  for (int tj = 0; tj < 4; ++tj) {
    v8f acc = {0.f, 0.f, 0.f, 0.f, 0.f, 0.f, 0.f, 0.f};
    for (int kc = 0; kc < 4; ++kc)
      acc = wmma_f16(load_frag(aw + kc * 512), load_frag(sB + (kc * 4 + tj) * 512), acc);
    int n = tj * 16 + nl;
    float bo = ob[n];
#pragma unroll
    for (int r = 0; r < 8; ++r) out[(tok0 + mb + r) * FEAT + n] = acc[r] + bo;
  }
}

// ---------------------------------------------------------------------------
// Host launcher
// ---------------------------------------------------------------------------
extern "C" void kernel_launch(void* const* d_in, const int* in_sizes, int n_in,
                              void* d_out, int out_size, void* d_ws, size_t ws_size,
                              hipStream_t stream) {
  const float* x    = (const float*)d_in[0];
  const float* c1w  = (const float*)d_in[1];
  const float* c1b  = (const float*)d_in[2];
  const float* c2w  = (const float*)d_in[3];
  const float* c2b  = (const float*)d_in[4];
  const float* fW1  = (const float*)d_in[5];
  const float* fb1  = (const float*)d_in[6];
  const float* fW2  = (const float*)d_in[7];
  const float* fb2  = (const float*)d_in[8];
  const float* cW1  = (const float*)d_in[9];
  const float* cb1  = (const float*)d_in[10];
  const float* cW2  = (const float*)d_in[11];
  const float* cb2  = (const float*)d_in[12];
  const float* eWih = (const float*)d_in[13];
  const float* eWhh = (const float*)d_in[14];
  const float* ebih = (const float*)d_in[15];
  const float* ebhh = (const float*)d_in[16];
  const float* aW1  = (const float*)d_in[17];
  const float* ab1  = (const float*)d_in[18];
  const float* aW2  = (const float*)d_in[19];
  const float* ab2  = (const float*)d_in[20];
  const float* bW   = (const float*)d_in[21];
  const float* bb   = (const float*)d_in[22];
  const float* h0W  = (const float*)d_in[23];
  const float* h0b  = (const float*)d_in[24];
  const float* c0W  = (const float*)d_in[25];
  const float* c0b  = (const float*)d_in[26];
  const float* dWih = (const float*)d_in[27];
  const float* dWhh = (const float*)d_in[28];
  const float* dbih = (const float*)d_in[29];
  const float* dbhh = (const float*)d_in[30];
  const float* oW   = (const float*)d_in[31];
  const float* ob   = (const float*)d_in[32];

  float* out0   = (float*)d_out;             // [B,S,F]
  float* out_tw = out0 + (size_t)TOK * FEAT; // [B,S,1]
  float* out_fw = out_tw + TOK;              // [B,S,F]

  // workspace carve-up
  char* p = (char*)d_ws;
  auto alloc = [&](size_t bytes) -> void* {
    void* r = (void*)p;
    p += (bytes + 255) & ~(size_t)255;
    return r;
  };
  float* gram  = (float*)alloc((size_t)BATCH * 4096 * 4);
  float* pool1 = (float*)alloc((size_t)BATCH * 16384 * 4);
  float* proc  = (float*)alloc((size_t)BATCH * 32 * 4);
  float* wx    = (float*)alloc((size_t)TOK * FEAT * 4);
  float* enc   = (float*)alloc((size_t)TOK * HID * 4);
  float* dec   = (float*)alloc((size_t)TOK * HID * 4);
  float* ts    = (float*)alloc((size_t)TOK * 4);
  float* ctx   = (float*)alloc((size_t)BATCH * HID * 4);
  float* bott  = (float*)alloc((size_t)BATCH * 32 * 4);
  float* dh0v  = (float*)alloc((size_t)BATCH * HID * 4);
  float* dc0v  = (float*)alloc((size_t)BATCH * HID * 4);
  float* zbase = (float*)alloc((size_t)BATCH * NG * 4);
  float* ebias = (float*)alloc(NG * 4);
  float* dbias = (float*)alloc(NG * 4);
  _Float16* eWihT = (_Float16*)alloc(64 * NG * 2);
  _Float16* eWhhT = (_Float16*)alloc(128 * NG * 2);
  _Float16* dWhhT = (_Float16*)alloc(128 * NG * 2);
  _Float16* oWT   = (_Float16*)alloc(128 * 64 * 2);
  _Float16* fW1T  = (_Float16*)alloc(64 * 32 * 2);
  _Float16* fW2T  = (_Float16*)alloc(32 * 64 * 2);
  _Float16* cW1T  = (_Float16*)alloc(96 * 64 * 2);
  _Float16* cW2T  = (_Float16*)alloc(64 * 64 * 2);

  prep_kernel<<<64, 256, 0, stream>>>(eWih, eWhh, dWhh, oW, fW1, fW2, cW1, cW2,
                                      ebih, ebhh, dbih, dbhh,
                                      eWihT, eWhhT, dWhhT, oWT, fW1T, fW2T, cW1T, cW2T,
                                      ebias, dbias);
  gram_kernel<<<BATCH, 128, 0, stream>>>(x, gram);
  conv1_kernel<<<BATCH, 256, 0, stream>>>(gram, c1w, c1b, pool1);
  conv2_kernel<<<BATCH, 256, 0, stream>>>(pool1, c2w, c2b, proc);
  feature_kernel<<<TOK / 64, 128, 0, stream>>>(x, proc, fW1T, fb1, fW2T, fb2,
                                               cW1T, cb1, cW2T, cb2, out_fw, wx);
  lstm_kernel<0><<<BATCH / 16, 256, 0, stream>>>(wx, eWihT, eWhhT, ebias,
                                                 nullptr, nullptr, nullptr, enc);
  tscore_kernel<<<TOK / 256, 256, 0, stream>>>(enc, aW1, ab1, aW2, ab2, ts);
  tctx_kernel<<<BATCH, 128, 0, stream>>>(ts, enc, out_tw, ctx);
  bott_kernel<<<BATCH, 128, 0, stream>>>(ctx, bW, bb, h0W, h0b, c0W, c0b, bott, dh0v, dc0v);
  dxp_kernel<<<(BATCH * NG + 255) / 256, 256, 0, stream>>>(bott, dWih, dbias, zbase);
  lstm_kernel<1><<<BATCH / 16, 256, 0, stream>>>(nullptr, nullptr, dWhhT, nullptr,
                                                 zbase, dh0v, dc0v, dec);
  outproj_kernel<<<TOK / 128, 256, 0, stream>>>(dec, oWT, ob, out0);
}
